// MultiHeadAttention_1340029796516
// MI455X (gfx1250) — matrix operand order repaired and checked by reading, not compiled, not run on hardware
//
#include <hip/hip_runtime.h>
#include <hip/hip_bf16.h>
#include <math.h>

typedef __bf16 bf16_t;
typedef __attribute__((ext_vector_type(16))) __bf16 v16bf;
typedef __attribute__((ext_vector_type(8)))  __bf16 v8bf;
typedef __attribute__((ext_vector_type(4)))  __bf16 v4bf;
typedef __attribute__((ext_vector_type(8)))  float  v8f;
typedef __attribute__((__vector_size__(4 * sizeof(int)))) int v4i;

#define D_MODEL   1024
#define NUM_HEADS 16
#define HEAD_DIM  64
#define BATCH     4
#define SEQ       2048
#define MTOT      (BATCH * SEQ)   // 8192 tokens

#if __has_builtin(__builtin_amdgcn_global_load_async_to_lds_b128)
#define HAVE_ASYNC_LDS 1
#else
#define HAVE_ASYNC_LDS 0
#endif

union AF { v16bf v; v8bf h[2]; };

static __device__ __forceinline__ v8f wmma_bf16(const AF& a, const AF& b, v8f c) {
    return __builtin_amdgcn_wmma_f32_16x16x32_bf16(
        /*neg_a=*/false, a.v, /*neg_b=*/false, b.v,
        /*c_mod=*/(short)0, c, /*reuse_a=*/false, /*reuse_b=*/false);
}

// 16-byte global -> LDS copy; async (ASYNCcnt) when the builtin exists.
static __device__ __forceinline__ void copy16_g2l(const bf16_t* g, bf16_t* l) {
#if HAVE_ASYNC_LDS
    __builtin_amdgcn_global_load_async_to_lds_b128((v4i*)g, (v4i*)l, 0, 0);
#else
    *(v8bf*)l = *(const v8bf*)g;
#endif
}

static __device__ __forceinline__ void wait_async_copies() {
#if HAVE_ASYNC_LDS
#if __has_builtin(__builtin_amdgcn_s_wait_asynccnt)
    __builtin_amdgcn_s_wait_asynccnt(0);
#else
    asm volatile("s_wait_asynccnt 0" ::: "memory");
#endif
#endif
}

// ---------------------------------------------------------------------------
// Elementwise fp32 -> bf16 (vectorized float4 -> 4x bf16).
// ---------------------------------------------------------------------------
__global__ __launch_bounds__(256)
void cvt_f32_bf16(const float* __restrict__ src, bf16_t* __restrict__ dst, int n4)
{
    const int i = blockIdx.x * 256 + threadIdx.x;
    if (i < n4) {
        const float4 v = ((const float4*)src)[i];
        v4bf o = { (bf16_t)v.x, (bf16_t)v.y, (bf16_t)v.z, (bf16_t)v.w };
        *(v4bf*)(dst + (size_t)i * 4) = o;
    }
}

// ---------------------------------------------------------------------------
// Projection GEMM: C[M=8192, N=1024] = A[M,K=1024] * W[N,K]^T + bias
// A, W are bf16 (pre-converted). Block tile 128x64, K-step 32, 8 waves;
// each wave owns a 32x32 C patch (4 WMMA accumulators, 4 WMMAs / K-step).
// Tiles staged to LDS with global_load_async_to_lds_b128 when available.
// mode 0: store bf16 row-major [tok, N]
// mode 1: store bf16 transposed-per-head V layout [b, h, dh, t]
// mode 2: store fp32 row-major (final output)
// ---------------------------------------------------------------------------
__global__ __launch_bounds__(256)
void proj_gemm(const bf16_t* __restrict__ Ab, const bf16_t* __restrict__ Wb,
               const float* __restrict__ bias,
               bf16_t* __restrict__ outb, float* __restrict__ outf, int mode)
{
    const int K = D_MODEL, N = D_MODEL;
    __shared__ __align__(16) bf16_t As[128 * 32];  // 8 KB
    __shared__ __align__(16) bf16_t Bs[64 * 32];   // 4 KB

    const int tid  = threadIdx.x;
    const int lane = tid & 31;
    const int wid  = tid >> 5;
    const int m0   = blockIdx.y * 128;
    const int n0   = blockIdx.x * 64;
    const int mw   = (wid & 3) * 32;   // wave's 32-row M strip
    const int nw   = (wid >> 2) * 32;  // wave's 32-col N strip
    const int lrow = lane & 15;
    const int lhi  = lane >> 4;        // 0 or 1

    v8f c00 = {0,0,0,0,0,0,0,0}, c01 = {0,0,0,0,0,0,0,0};
    v8f c10 = {0,0,0,0,0,0,0,0}, c11 = {0,0,0,0,0,0,0,0};

    for (int k0 = 0; k0 < K; k0 += 32) {
        __syncthreads();   // previous fragments consumed before overwrite
        // ---- stage A tile (128x32 bf16 = 512 x 16B chunks, 2/thread) ----
        #pragma unroll
        for (int i = 0; i < 2; ++i) {
            const int ch  = tid + i * 256;
            const int row = ch >> 2, off = (ch & 3) * 8;
            copy16_g2l(Ab + (size_t)(m0 + row) * K + k0 + off, As + row * 32 + off);
        }
        // ---- stage W tile (64x32 bf16 = 256 x 16B chunks, 1/thread) ----
        {
            const int row = tid >> 2, off = (tid & 3) * 8;
            copy16_g2l(Wb + (size_t)(n0 + row) * K + k0 + off, Bs + row * 32 + off);
        }
        wait_async_copies();
        __syncthreads();

        // ---- fragments ----
        AF a0, a1, b0, b1;
        const bf16_t* ar0 = As + (mw + lrow) * 32;
        const bf16_t* ar1 = As + (mw + 16 + lrow) * 32;
        a0.h[0] = *(const v8bf*)(ar0 + lhi * 16);
        a0.h[1] = *(const v8bf*)(ar0 + lhi * 16 + 8);
        a1.h[0] = *(const v8bf*)(ar1 + lhi * 16);
        a1.h[1] = *(const v8bf*)(ar1 + lhi * 16 + 8);
        const bf16_t* br0 = Bs + (nw + lrow) * 32 + lhi * 16;
        const bf16_t* br1 = Bs + (nw + 16 + lrow) * 32 + lhi * 16;
        b0.h[0] = *(const v8bf*)(br0);  b0.h[1] = *(const v8bf*)(br0 + 8);
        b1.h[0] = *(const v8bf*)(br1);  b1.h[1] = *(const v8bf*)(br1 + 8);

        c00 = wmma_bf16(a0, b0, c00);
        c01 = wmma_bf16(a0, b1, c01);
        c10 = wmma_bf16(a1, b0, c10);
        c11 = wmma_bf16(a1, b1, c11);
    }

    // ---- epilogue: bias + store ----
    #pragma unroll
    for (int mi = 0; mi < 2; ++mi) {
        #pragma unroll
        for (int nt = 0; nt < 2; ++nt) {
            v8f c = mi ? (nt ? c11 : c10) : (nt ? c01 : c00);
            const int col = n0 + nw + nt * 16 + lrow;
            const float bv = bias[col];
            #pragma unroll
            for (int r = 0; r < 8; ++r) {
                const int mrow = m0 + mw + mi * 16 + r + lhi * 8;
                const float val = c[r] + bv;
                if (mode == 0) {
                    outb[(size_t)mrow * N + col] = (bf16_t)val;
                } else if (mode == 1) {
                    const int b = mrow >> 11, t = mrow & (SEQ - 1);
                    const int h = col >> 6,  dh = col & 63;
                    outb[((size_t)((b * NUM_HEADS + h) * HEAD_DIM + dh)) * SEQ + t] = (bf16_t)val;
                } else {
                    outf[(size_t)mrow * N + col] = val;
                }
            }
        }
    }
}

// ---------------------------------------------------------------------------
// Flash attention: one wave per 16-query tile per (b,h).
// Q/K row-major bf16 [tok, 1024]; V transposed bf16 [b,h,dh,t].
// Online softmax with running row max/sum; P staged through LDS to convert
// C-layout (lane=col) scores into A-layout (lane=row) fragments.
// ---------------------------------------------------------------------------
__global__ __launch_bounds__(256)
void flash_attn(const bf16_t* __restrict__ Qp, const bf16_t* __restrict__ Kp,
                const bf16_t* __restrict__ VpT, bf16_t* __restrict__ AO)
{
    __shared__ __align__(16) bf16_t Pbuf[8][16 * 32];

    const int tid  = threadIdx.x;
    const int lane = tid & 31;
    const int wid  = tid >> 5;
    const int tile = blockIdx.x * 8 + wid;     // 0..8191
    const int qt   = tile & 127;               // SEQ/16 = 128 tiles
    const int bh   = tile >> 7;
    const int h    = bh & (NUM_HEADS - 1);
    const int b    = bh >> 4;
    const int qbase = b * SEQ + qt * 16;
    const int lrow = lane & 15;
    const int lhi  = lane >> 4;

    // Q A-fragments (d 0..31 and 32..63), pinned for the whole loop
    AF qa0, qa1;
    {
        const bf16_t* qr = Qp + (size_t)(qbase + lrow) * D_MODEL + h * HEAD_DIM;
        qa0.h[0] = *(const v8bf*)(qr + lhi * 16);
        qa0.h[1] = *(const v8bf*)(qr + lhi * 16 + 8);
        qa1.h[0] = *(const v8bf*)(qr + 32 + lhi * 16);
        qa1.h[1] = *(const v8bf*)(qr + 32 + lhi * 16 + 8);
    }

    v8f o0 = {0,0,0,0,0,0,0,0}, o1 = {0,0,0,0,0,0,0,0};
    v8f o2 = {0,0,0,0,0,0,0,0}, o3 = {0,0,0,0,0,0,0,0};
    float mrow[8], lsum[8];
    #pragma unroll
    for (int r = 0; r < 8; ++r) { mrow[r] = -INFINITY; lsum[r] = 0.0f; }

    const float SCALE = 0.125f;  // 1/sqrt(64)
    bf16_t* myP = Pbuf[wid];
    const bf16_t* vbase = VpT + ((size_t)((b * NUM_HEADS + h) * HEAD_DIM)) * SEQ;

    for (int kc = 0; kc < SEQ; kc += 32) {
        // ---- K B-fragments: key tiles j=0/1, d-halves 0/1 ----
        const bf16_t* krb = Kp + (size_t)(b * SEQ + kc) * D_MODEL + h * HEAD_DIM;
        const bf16_t* kr0 = krb + (size_t)lrow * D_MODEL;
        const bf16_t* kr1 = krb + (size_t)(16 + lrow) * D_MODEL;
        if (kc + 32 < SEQ)
            __builtin_prefetch(krb + (size_t)32 * D_MODEL, 0, 1);  // global_prefetch_b8

        AF kb00, kb01, kb10, kb11;
        kb00.h[0] = *(const v8bf*)(kr0 + lhi * 16);
        kb00.h[1] = *(const v8bf*)(kr0 + lhi * 16 + 8);
        kb01.h[0] = *(const v8bf*)(kr0 + 32 + lhi * 16);
        kb01.h[1] = *(const v8bf*)(kr0 + 32 + lhi * 16 + 8);
        kb10.h[0] = *(const v8bf*)(kr1 + lhi * 16);
        kb10.h[1] = *(const v8bf*)(kr1 + lhi * 16 + 8);
        kb11.h[0] = *(const v8bf*)(kr1 + 32 + lhi * 16);
        kb11.h[1] = *(const v8bf*)(kr1 + 32 + lhi * 16 + 8);

        v8f s0 = {0,0,0,0,0,0,0,0}, s1 = {0,0,0,0,0,0,0,0};
        s0 = wmma_bf16(qa0, kb00, s0);
        s0 = wmma_bf16(qa1, kb01, s0);
        s1 = wmma_bf16(qa0, kb10, s1);
        s1 = wmma_bf16(qa1, kb11, s1);

        // ---- online softmax over this 32-key chunk ----
        float fscale[8];
        #pragma unroll
        for (int r = 0; r < 8; ++r) {
            float a0 = s0[r] * SCALE, a1 = s1[r] * SCALE;
            float mx = fmaxf(a0, a1);
            mx = fmaxf(mx, __shfl_xor(mx, 8, 32));
            mx = fmaxf(mx, __shfl_xor(mx, 4, 32));
            mx = fmaxf(mx, __shfl_xor(mx, 2, 32));
            mx = fmaxf(mx, __shfl_xor(mx, 1, 32));
            const float mnew = fmaxf(mrow[r], mx);
            const float f  = __expf(mrow[r] - mnew);
            const float p0 = __expf(a0 - mnew);
            const float p1 = __expf(a1 - mnew);
            float rs = p0 + p1;
            rs += __shfl_xor(rs, 8, 32);
            rs += __shfl_xor(rs, 4, 32);
            rs += __shfl_xor(rs, 2, 32);
            rs += __shfl_xor(rs, 1, 32);
            lsum[r] = lsum[r] * f + rs;
            mrow[r] = mnew;
            fscale[r] = f;
            // stage P in A-matrix memory order: row = r + lhi*8, col = lrow (+16)
            myP[(r + lhi * 8) * 32 + lrow]      = (bf16_t)p0;
            myP[(r + lhi * 8) * 32 + lrow + 16] = (bf16_t)p1;
        }
        #pragma unroll
        for (int r = 0; r < 8; ++r) {
            o0[r] *= fscale[r]; o1[r] *= fscale[r];
            o2[r] *= fscale[r]; o3[r] *= fscale[r];
        }

        asm volatile("s_wait_dscnt 0" ::: "memory");  // P stores visible before A-frag reload

        // ---- P A-fragment (16x32 over this key chunk) ----
        AF pa;
        const bf16_t* pr = myP + lrow * 32;
        pa.h[0] = *(const v8bf*)(pr + lhi * 16);
        pa.h[1] = *(const v8bf*)(pr + lhi * 16 + 8);

        // ---- V B-fragments: d-tiles 0..3, keys contiguous (transposed V) ----
        const bf16_t* vk = vbase + kc + lhi * 16;
        AF vb0, vb1, vb2, vb3;
        { const bf16_t* p = vk + (size_t)(lrow) * SEQ;
          vb0.h[0] = *(const v8bf*)p; vb0.h[1] = *(const v8bf*)(p + 8); }
        { const bf16_t* p = vk + (size_t)(16 + lrow) * SEQ;
          vb1.h[0] = *(const v8bf*)p; vb1.h[1] = *(const v8bf*)(p + 8); }
        { const bf16_t* p = vk + (size_t)(32 + lrow) * SEQ;
          vb2.h[0] = *(const v8bf*)p; vb2.h[1] = *(const v8bf*)(p + 8); }
        { const bf16_t* p = vk + (size_t)(48 + lrow) * SEQ;
          vb3.h[0] = *(const v8bf*)p; vb3.h[1] = *(const v8bf*)(p + 8); }

        o0 = wmma_bf16(pa, vb0, o0);
        o1 = wmma_bf16(pa, vb1, o1);
        o2 = wmma_bf16(pa, vb2, o2);
        o3 = wmma_bf16(pa, vb3, o3);
    }

    // ---- finalize: divide by row sums, store bf16 row-major ----
    #pragma unroll
    for (int r = 0; r < 8; ++r) {
        const float inv = 1.0f / lsum[r];
        const int row = qbase + r + lhi * 8;
        bf16_t* dst = AO + (size_t)row * D_MODEL + h * HEAD_DIM + lrow;
        dst[0]  = (bf16_t)(o0[r] * inv);
        dst[16] = (bf16_t)(o1[r] * inv);
        dst[32] = (bf16_t)(o2[r] * inv);
        dst[48] = (bf16_t)(o3[r] * inv);
    }
}

// ---------------------------------------------------------------------------
extern "C" void kernel_launch(void* const* d_in, const int* in_sizes, int n_in,
                              void* d_out, int out_size, void* d_ws, size_t ws_size,
                              hipStream_t stream) {
    (void)in_sizes; (void)n_in; (void)out_size; (void)ws_size;
    const float* q  = (const float*)d_in[0];
    const float* k  = (const float*)d_in[1];
    const float* v  = (const float*)d_in[2];
    const float* wq = (const float*)d_in[3];
    const float* bq = (const float*)d_in[4];
    const float* wk = (const float*)d_in[5];
    const float* bk = (const float*)d_in[6];
    const float* wv = (const float*)d_in[7];
    const float* bv = (const float*)d_in[8];
    const float* wo = (const float*)d_in[9];
    const float* bo = (const float*)d_in[10];
    float* out = (float*)d_out;

    char* w = (char*)d_ws;
    const size_t seg  = (size_t)MTOT * D_MODEL * sizeof(bf16_t);     // 16 MB
    const size_t wseg = (size_t)D_MODEL * D_MODEL * sizeof(bf16_t);  //  2 MB
    bf16_t* Xq  = (bf16_t*)(w);
    bf16_t* Xk  = (bf16_t*)(w + seg);
    bf16_t* Xv  = (bf16_t*)(w + 2 * seg);
    bf16_t* Wq  = (bf16_t*)(w + 3 * seg);
    bf16_t* Wk  = (bf16_t*)(w + 3 * seg + wseg);
    bf16_t* Wv  = (bf16_t*)(w + 3 * seg + 2 * wseg);
    bf16_t* Wo  = (bf16_t*)(w + 3 * seg + 3 * wseg);
    bf16_t* Qp  = (bf16_t*)(w + 3 * seg + 4 * wseg);
    bf16_t* Kp  = (bf16_t*)(w + 4 * seg + 4 * wseg);
    bf16_t* VpT = (bf16_t*)(w + 5 * seg + 4 * wseg);
    bf16_t* AO  = (bf16_t*)(w + 6 * seg + 4 * wseg);   // total 120 MB

    const int nX4 = MTOT * D_MODEL / 4;      // 2,097,152
    const int nW4 = D_MODEL * D_MODEL / 4;   //   262,144
    dim3 blk(256);
    cvt_f32_bf16<<<dim3(nX4 / 256), blk, 0, stream>>>(q,  Xq, nX4);
    cvt_f32_bf16<<<dim3(nX4 / 256), blk, 0, stream>>>(k,  Xk, nX4);
    cvt_f32_bf16<<<dim3(nX4 / 256), blk, 0, stream>>>(v,  Xv, nX4);
    cvt_f32_bf16<<<dim3(nW4 / 256), blk, 0, stream>>>(wq, Wq, nW4);
    cvt_f32_bf16<<<dim3(nW4 / 256), blk, 0, stream>>>(wk, Wk, nW4);
    cvt_f32_bf16<<<dim3(nW4 / 256), blk, 0, stream>>>(wv, Wv, nW4);
    cvt_f32_bf16<<<dim3(nW4 / 256), blk, 0, stream>>>(wo, Wo, nW4);

    dim3 grid(D_MODEL / 64, MTOT / 128);  // (16, 64)
    proj_gemm<<<grid, blk, 0, stream>>>(Xq, Wq, bq, Qp,  nullptr, 0);
    proj_gemm<<<grid, blk, 0, stream>>>(Xk, Wk, bk, Kp,  nullptr, 0);
    proj_gemm<<<grid, blk, 0, stream>>>(Xv, Wv, bv, VpT, nullptr, 1);
    flash_attn<<<dim3(MTOT / 16 / 8), blk, 0, stream>>>(Qp, Kp, VpT, AO);
    proj_gemm<<<grid, blk, 0, stream>>>(AO, Wo, bo, nullptr, out, 2);
}
